// MARN_cell_20504173871336
// MI455X (gfx1250) — matrix operand order repaired and checked by reading, not compiled, not run on hardware
//
#include <hip/hip_runtime.h>
#include <hip/hip_bf16.h>
#include <math.h>

// ---------------------------------------------------------------------------
// MARN cell for MI455X (gfx1250): sequential scan over T=128 steps.
// All GEMMs run through v_wmma_f32_16x16x32_bf16 (bf16 A/B, f32 accumulate).
// All GEMM operands are pre-converted bf16, zero-padded to K%32==0, and the
// B matrices are packed in WMMA fragment order so the inner loop is pure
// vector-load + wmma.
// ---------------------------------------------------------------------------

#define DHv    128
#define GAT    512          // 4*DH
#define NBAT   1024
#define TSTEPS 128
#define DLDIM  300
#define KLPAD  320          // x_l K padded to 32
#define DADIM  74
#define KAPAD  96           // x_a K padded to 32

typedef __attribute__((ext_vector_type(16))) __bf16 v16bf;
typedef __attribute__((ext_vector_type(8)))  __bf16 v8bf;
typedef __attribute__((ext_vector_type(8)))  float  v8f;

__device__ __forceinline__ float sigm(float x) { return 1.f / (1.f + __expf(-x)); }

// ---------------------------------------------------------------------------
// WMMA GEMM:  C(1024, 512) = sum_s A_s(1024, Kpad_s) @ B_s + bias
//   A_s : bf16 row-major [N][Kpad]   (zero-padded, Kpad % 32 == 0)
//   B_s : bf16 packed fragment order [G/16][Kpad/32][lane*16+i]
// One wave computes a 16x64 strip (4 j-tiles): 1 A fragment feeds 4 WMMAs.
// Fragment layouts used here (A/B use contiguous k, not the ISA table order; C as in ISA):
//   A 16x32 bf16: lane&15 = row; lanes<16 hold K {0..15},
//                 lanes>=16 hold K {16..31}  -> two 16B contiguous loads.
//   B 32x16 bf16: lane&15 = col; K = i + 16*(lane>>4) -> packed contiguous 32B.
//   C 16x16 f32 : vgpr r, lane l -> M = r + 8*(l>>4), N = l&15.
// ---------------------------------------------------------------------------
struct GemmArgs {
    const __bf16* A[4];
    const __bf16* B[4];
    int K[4];            // padded K, multiple of 32
    int nsrc;
    const float* bias;   // length 512 (combined)
    float* C;            // [1024][512]
};

__global__ __launch_bounds__(128) void gemm_bf16_wmma(GemmArgs ga) {
    const int lane  = threadIdx.x & 31;
    const int wave  = threadIdx.x >> 5;          // 0..3 : n-tile within block
    const int whalf = lane >> 4;
    const int mrow  = lane & 15;
    const int tile_n = (blockIdx.y * 4 + wave) * 16;
    const int j0     = blockIdx.x * 64;          // 4 consecutive 16-col tiles

    v8f acc0, acc1, acc2, acc3;
    {
        const float b0 = ga.bias[j0 + mrow];
        const float b1 = ga.bias[j0 + 16 + mrow];
        const float b2 = ga.bias[j0 + 32 + mrow];
        const float b3 = ga.bias[j0 + 48 + mrow];
#pragma unroll
        for (int r = 0; r < 8; ++r) { acc0[r] = b0; acc1[r] = b1; acc2[r] = b2; acc3[r] = b3; }
    }

    for (int s = 0; s < ga.nsrc; ++s) {
        const int Kp = ga.K[s];
        const int nks = Kp >> 5;
        const __bf16* __restrict__ arow = ga.A[s] + (size_t)(tile_n + mrow) * Kp;
        const __bf16* __restrict__ bbas = ga.B[s] + (size_t)lane * 16;
        for (int ks = 0; ks < nks; ++ks) {
            const v8bf alo = *(const v8bf*)(arow + ks * 32 + whalf * 16);
            const v8bf ahi = *(const v8bf*)(arow + ks * 32 + whalf * 16 + 8);
            const v16bf af = __builtin_shufflevector(alo, ahi,
                0, 1, 2, 3, 4, 5, 6, 7, 8, 9, 10, 11, 12, 13, 14, 15);
#pragma unroll
            for (int c = 0; c < 4; ++c) {
                const __bf16* bp = bbas + ((size_t)((j0 >> 4) + c) * nks + ks) * 512;
                const v8bf blo = *(const v8bf*)(bp);
                const v8bf bhi = *(const v8bf*)(bp + 8);
                const v16bf bfr = __builtin_shufflevector(blo, bhi,
                    0, 1, 2, 3, 4, 5, 6, 7, 8, 9, 10, 11, 12, 13, 14, 15);
                v8f* accp = (c == 0) ? &acc0 : (c == 1) ? &acc1 : (c == 2) ? &acc2 : &acc3;
                *accp = __builtin_amdgcn_wmma_f32_16x16x32_bf16(
                    false, af, false, bfr, (short)0, *accp, false, false);
            }
        }
    }

    float* crow = ga.C + (size_t)tile_n * GAT + j0;
#pragma unroll
    for (int r = 0; r < 8; ++r) {
        const size_t row = (size_t)(r + 8 * whalf) * GAT;
        crow[row + mrow]      = acc0[r];
        crow[row + 16 + mrow] = acc1[r];
        crow[row + 32 + mrow] = acc2[r];
        crow[row + 48 + mrow] = acc3[r];
    }
}

// ---------------------------------------------------------------------------
// Per-step stable partition of the batch by speaker (LDS prefix scan).
// ---------------------------------------------------------------------------
__global__ __launch_bounds__(NBAT) void perm_kernel(const float* __restrict__ qm_t,
                                                    int* qm_idx, int* perm0,
                                                    int* perm1, int* n0_out) {
    __shared__ int buf[NBAT];
    const int tid = threadIdx.x;
    const int bit = (qm_t[tid * 2 + 1] > qm_t[tid * 2 + 0]) ? 1 : 0;
    qm_idx[tid] = bit;
    buf[tid]    = bit;
    __syncthreads();
    for (int off = 1; off < NBAT; off <<= 1) {
        const int add = (tid >= off) ? buf[tid - off] : 0;
        __syncthreads();
        buf[tid] += add;
        __syncthreads();
    }
    const int ones_incl  = buf[tid];
    const int total_ones = buf[NBAT - 1];
    const int ones_excl  = ones_incl - bit;
    const int zeros_excl = tid - ones_excl;
    const int n0         = NBAT - total_ones;
    perm0[bit ? (n0 + ones_excl) : zeros_excl]         = tid;  // zeros first
    perm1[bit ? ones_excl : (total_ones + zeros_excl)] = tid;  // ones first
    if (tid == 0) *n0_out = n0;
}

// q0_sel = q_sel[perm0] * (party==0);  q1_sel = q_sel[perm1] * (party==1)
// Writes f32 (for compose) and bf16 mirrors (for the WMMA GEMMs).
__global__ void gather_sel_kernel(const float* __restrict__ q,
                                  const int* __restrict__ qm_idx,
                                  const int* __restrict__ perm0,
                                  const int* __restrict__ perm1,
                                  float* q0_sel, float* q1_sel,
                                  __bf16* q0_sel_b, __bf16* q1_sel_b) {
    const int gid = blockIdx.x * blockDim.x + threadIdx.x;
    if (gid >= NBAT * DHv) return;
    const int i = gid >> 7, d = gid & (DHv - 1);
    const int p0 = perm0[i], b0 = qm_idx[p0];
    const int p1 = perm1[i], b1 = qm_idx[p1];
    const float s0 = q[(size_t)p0 * 2 * DHv + (size_t)b0 * DHv + d];
    const float s1 = q[(size_t)p1 * 2 * DHv + (size_t)b1 * DHv + d];
    const float v0 = (b0 == 0) ? s0 : 0.f;
    const float v1 = (b1 == 1) ? s1 : 0.f;
    q0_sel[gid] = v0;  q0_sel_b[gid] = (__bf16)v0;
    q1_sel[gid] = v1;  q1_sel_b[gid] = (__bf16)v1;
}

// LSTMCell gates (PyTorch order i,f,g,o); conditional state update on count>0
__global__ void lstm_gates_kernel(const float* __restrict__ gates,
                                  float* h, float* c, __bf16* h_b,
                                  const int* __restrict__ n0p, int mode) {
    const int gid = blockIdx.x * blockDim.x + threadIdx.x;
    if (gid >= NBAT * DHv) return;
    const int n0  = *n0p;
    const int cnt = (mode == 0) ? n0 : (NBAT - n0);
    if (cnt <= 0) return;
    const int n = gid >> 7, d = gid & (DHv - 1);
    const float* g = gates + (size_t)n * GAT;
    const float ig = sigm(g[d]);
    const float fg = sigm(g[DHv + d]);
    const float gg = tanhf(g[2 * DHv + d]);
    const float og = sigm(g[3 * DHv + d]);
    const float c2 = fg * c[gid] + ig * gg;
    const float h2 = og * tanhf(c2);
    c[gid] = c2;
    h[gid] = h2;
    h_b[gid] = (__bf16)h2;
}

// h_q / h_0 composition, q_new update, write h_q slice of the output
__global__ void compose_kernel(const float* __restrict__ h_q0,
                               const float* __restrict__ h_q1,
                               const float* __restrict__ q0_sel,
                               const float* __restrict__ q1_sel,
                               const int* __restrict__ n0p,
                               const float* __restrict__ qm_t,
                               float* h_q, __bf16* h_q_b, float* q, float* out_t) {
    const int gid = blockIdx.x * blockDim.x + threadIdx.x;
    if (gid >= NBAT * DHv) return;
    const int n = gid >> 7, d = gid & (DHv - 1);
    const int n0 = *n0p;
    int tail = n - n0;
    tail = (tail < 0) ? 0 : ((tail > NBAT - 1) ? NBAT - 1 : tail);
    const bool front = (n < n0);
    const float hq = front ? h_q0[gid] : h_q1[(size_t)tail * DHv + d];
    const float h0 = front ? q0_sel[gid] : q1_sel[(size_t)tail * DHv + d];
    const float qm0 = qm_t[n * 2 + 0], qm1 = qm_t[n * 2 + 1];
    q[(size_t)n * 2 * DHv + d]       = h0 * (1.f - qm0) + hq * qm0;
    q[(size_t)n * 2 * DHv + DHv + d] = h0 * (1.f - qm1) + hq * qm1;
    h_q[gid]   = hq;
    h_q_b[gid] = (__bf16)hq;
    out_t[(size_t)n * GAT + 3 * DHv + d] = hq;
}

// LSTHM gates (order f,i,o,ch); writes state + bf16 mirror + output slice
__global__ void lsthm_gates_kernel(const float* __restrict__ sums,
                                   float* c, float* h, __bf16* h_b,
                                   float* out_t, int out_off) {
    const int gid = blockIdx.x * blockDim.x + threadIdx.x;
    if (gid >= NBAT * DHv) return;
    const int n = gid >> 7, d = gid & (DHv - 1);
    const float* s = sums + (size_t)n * GAT;
    const float f  = sigm(s[d]);
    const float i  = sigm(s[DHv + d]);
    const float o  = sigm(s[2 * DHv + d]);
    const float ch = tanhf(s[3 * DHv + d]);
    const float c2 = f * c[gid] + i * ch;
    const float h2 = tanhf(c2) * o;
    c[gid] = c2;
    h[gid] = h2;
    h_b[gid] = (__bf16)h2;
    out_t[(size_t)n * GAT + out_off + d] = h2;
}

// Rank-1 cross attention: z[n,i] = softmax_m(c_l[n,i]*coef[n]*Wk[m]) . c_a[n,:]
__global__ __launch_bounds__(DHv) void attention_kernel(
        const float* __restrict__ c_l, const float* __restrict__ c_a,
        const float* __restrict__ Wq,  const float* __restrict__ Wk,
        float* z_l, __bf16* z_l_b, float* out_t) {
    __shared__ float ca[DHv], wk[DHv], red[DHv];
    const int n = blockIdx.x, i = threadIdx.x;
    const float cav = c_a[(size_t)n * DHv + i];
    ca[i]  = cav;
    wk[i]  = Wk[i];
    red[i] = cav * Wq[i];
    __syncthreads();
    for (int off = DHv / 2; off > 0; off >>= 1) {
        if (i < off) red[i] += red[i + off];
        __syncthreads();
    }
    const float coef = red[0] * 0.08838834764831845f;  // 1/sqrt(128)
    const float a = c_l[(size_t)n * DHv + i] * coef;
    float mx = -INFINITY;
    for (int m = 0; m < DHv; ++m) mx = fmaxf(mx, a * wk[m]);
    float den = 0.f, num = 0.f;
    for (int m = 0; m < DHv; ++m) {
        const float e = __expf(a * wk[m] - mx);
        den += e;
        num += e * ca[m];
    }
    const float z = num / den;
    z_l[(size_t)n * DHv + i]             = z;
    z_l_b[(size_t)n * DHv + i]           = (__bf16)z;
    out_t[(size_t)n * GAT + 2 * DHv + i] = z;
}

// Pack f32 weights into bf16 WMMA B-fragment order:
//   dst[jt][ks][lane*16+i]  with  k = ks*32 + i + 16*(lane>>4), j = jt*16 + (lane&15)
__global__ void pack_weight_kernel(const float* __restrict__ src, __bf16* dst,
                                   int K, int Kpad, int trans) {
    const int gid = blockIdx.x * blockDim.x + threadIdx.x;
    const int total = (GAT / 16) * (Kpad / 32) * 512;
    if (gid >= total) return;
    const int e    = gid & 511;           // lane*16 + i
    const int rest = gid >> 9;
    const int nks  = Kpad >> 5;
    const int ks   = rest % nks;
    const int jt   = rest / nks;
    const int lane = e >> 4, i = e & 15;
    const int k = ks * 32 + i + 16 * (lane >> 4);
    const int j = jt * 16 + (lane & 15);
    float v = 0.f;
    if (k < K) v = trans ? src[(size_t)j * K + k] : src[(size_t)k * GAT + j];
    dst[gid] = (__bf16)v;
}

// Per-step conversion of x_l / x_a rows into zero-padded bf16 [N][Kpad]
__global__ void convert_x_kernel(const float* __restrict__ src, __bf16* dst,
                                 int K, int Kpad) {
    const int gid = blockIdx.x * blockDim.x + threadIdx.x;
    if (gid >= NBAT * Kpad) return;
    const int n = gid / Kpad, k = gid - n * Kpad;
    dst[gid] = (__bf16)((k < K) ? src[(size_t)n * K + k] : 0.f);
}

__global__ void combine_bias_kernel(float* dst, const float* a, const float* b,
                                    const float* c, const float* d) {
    const int j = blockIdx.x * blockDim.x + threadIdx.x;
    if (j >= GAT) return;
    float v = a[j] + b[j];
    if (c) v += c[j];
    if (d) v += d[j];
    dst[j] = v;
}

__global__ void zero_words_kernel(unsigned int* p, int n) {
    const int gid = blockIdx.x * blockDim.x + threadIdx.x;
    if (gid < n) p[gid] = 0u;
}

// ---------------------------------------------------------------------------
// Host side
// ---------------------------------------------------------------------------
static void launch_gemm(hipStream_t s, float* C, const float* bias,
                        const __bf16* A0, const __bf16* B0, int K0,
                        const __bf16* A1, const __bf16* B1, int K1,
                        const __bf16* A2, const __bf16* B2, int K2,
                        const __bf16* A3, const __bf16* B3, int K3, int nsrc) {
    GemmArgs ga;
    ga.A[0] = A0; ga.A[1] = A1; ga.A[2] = A2; ga.A[3] = A3;
    ga.B[0] = B0; ga.B[1] = B1; ga.B[2] = B2; ga.B[3] = B3;
    ga.K[0] = K0; ga.K[1] = K1; ga.K[2] = K2; ga.K[3] = K3;
    ga.nsrc = nsrc; ga.bias = bias; ga.C = C;
    gemm_bf16_wmma<<<dim3(GAT / 64, NBAT / 64), 128, 0, s>>>(ga);
}

extern "C" void kernel_launch(void* const* d_in, const int* in_sizes, int n_in,
                              void* d_out, int out_size, void* d_ws, size_t ws_size,
                              hipStream_t stream) {
    // setup_inputs() insertion order, params flattened depth-first:
    //  0 x  1 x_l  2 x_a  3 qmask
    //  4..11  lsthm_l  W.w W.b U.w U.b V.w V.b S.w S.b
    // 12..19  lsthm_a  (same)
    // 20..23  lstm_q0  w_ih w_hh b_ih b_hh
    // 24..27  lstm_q1  (same)
    // 28 Wq   29 Wk
    const float* x_l   = (const float*)d_in[1];
    const float* x_a   = (const float*)d_in[2];
    const float* qmask = (const float*)d_in[3];
    const float* Wl_w = (const float*)d_in[4];  const float* Wl_b = (const float*)d_in[5];
    const float* Ul_w = (const float*)d_in[6];  const float* Ul_b = (const float*)d_in[7];
    const float* Vl_w = (const float*)d_in[8];  const float* Vl_b = (const float*)d_in[9];
    const float* Sl_w = (const float*)d_in[10]; const float* Sl_b = (const float*)d_in[11];
    const float* Wa_w = (const float*)d_in[12]; const float* Wa_b = (const float*)d_in[13];
    const float* Ua_w = (const float*)d_in[14]; const float* Ua_b = (const float*)d_in[15];
    const float* Va_w = (const float*)d_in[16]; const float* Va_b = (const float*)d_in[17];
    const float* Sa_w = (const float*)d_in[18]; const float* Sa_b = (const float*)d_in[19];
    const float* q0_wih = (const float*)d_in[20]; const float* q0_whh = (const float*)d_in[21];
    const float* q0_bih = (const float*)d_in[22]; const float* q0_bhh = (const float*)d_in[23];
    const float* q1_wih = (const float*)d_in[24]; const float* q1_whh = (const float*)d_in[25];
    const float* q1_bih = (const float*)d_in[26]; const float* q1_bhh = (const float*)d_in[27];
    const float* Wq = (const float*)d_in[28];
    const float* Wk = (const float*)d_in[29];
    float* out = (float*)d_out;

    // -------- workspace bump allocator (256B aligned regions) --------
    char*  ws  = (char*)d_ws;
    size_t off = 0;
    auto alloc = [&](size_t bytes) -> char* {
        char* p = ws + off;
        off = (off + bytes + 255) & ~(size_t)255;
        return p;
    };
    const size_t ND = (size_t)NBAT * DHv;

    float* states = (float*)alloc(11 * ND * sizeof(float));  // contiguous, zeroed
    float* h_l  = states + 0 * ND;
    float* h_a  = states + 1 * ND;
    float* h_q0 = states + 2 * ND;
    float* h_q1 = states + 3 * ND;
    float* c_l  = states + 4 * ND;
    float* c_a  = states + 5 * ND;
    float* c_q0 = states + 6 * ND;
    float* c_q1 = states + 7 * ND;
    float* z_l  = states + 8 * ND;
    float* q    = states + 9 * ND;   // (N,2,DH) -> 2*ND

    // persistent bf16 mirrors (zeroed like the states)
    __bf16* mirrors = (__bf16*)alloc(5 * ND * sizeof(__bf16));
    __bf16* h_q0_b = mirrors + 0 * ND;
    __bf16* h_q1_b = mirrors + 1 * ND;
    __bf16* h_l_b  = mirrors + 2 * ND;
    __bf16* h_a_b  = mirrors + 3 * ND;
    __bf16* z_l_b  = mirrors + 4 * ND;

    // per-step scratch (fully rewritten each step before use)
    float*  h_q      = (float*)alloc(ND * sizeof(float));
    __bf16* h_q_b    = (__bf16*)alloc(ND * sizeof(__bf16));
    float*  q0_sel   = (float*)alloc(ND * sizeof(float));
    float*  q1_sel   = (float*)alloc(ND * sizeof(float));
    __bf16* q0_sel_b = (__bf16*)alloc(ND * sizeof(__bf16));
    __bf16* q1_sel_b = (__bf16*)alloc(ND * sizeof(__bf16));
    __bf16* xl_b     = (__bf16*)alloc((size_t)NBAT * KLPAD * sizeof(__bf16));
    __bf16* xa_b     = (__bf16*)alloc((size_t)NBAT * KAPAD * sizeof(__bf16));
    float* gates_q0 = (float*)alloc((size_t)NBAT * GAT * sizeof(float));
    float* gates_q1 = (float*)alloc((size_t)NBAT * GAT * sizeof(float));
    float* sums_l   = (float*)alloc((size_t)NBAT * GAT * sizeof(float));
    float* sums_a   = (float*)alloc((size_t)NBAT * GAT * sizeof(float));
    int* qm_idx = (int*)alloc(NBAT * sizeof(int));
    int* perm0  = (int*)alloc(NBAT * sizeof(int));
    int* perm1  = (int*)alloc(NBAT * sizeof(int));
    int* n0_ptr = (int*)alloc(sizeof(int));

    // packed bf16 weights (WMMA fragment order), size = 512*Kpad each
    __bf16* Bq0_ih = (__bf16*)alloc((size_t)DHv * GAT * sizeof(__bf16));
    __bf16* Bq0_hh = (__bf16*)alloc((size_t)DHv * GAT * sizeof(__bf16));
    __bf16* Bq1_ih = (__bf16*)alloc((size_t)DHv * GAT * sizeof(__bf16));
    __bf16* Bq1_hh = (__bf16*)alloc((size_t)DHv * GAT * sizeof(__bf16));
    __bf16* BlW = (__bf16*)alloc((size_t)KLPAD * GAT * sizeof(__bf16));
    __bf16* BlU = (__bf16*)alloc((size_t)DHv * GAT * sizeof(__bf16));
    __bf16* BlV = (__bf16*)alloc((size_t)DHv * GAT * sizeof(__bf16));
    __bf16* BlS = (__bf16*)alloc((size_t)DHv * GAT * sizeof(__bf16));
    __bf16* BaW = (__bf16*)alloc((size_t)KAPAD * GAT * sizeof(__bf16));
    __bf16* BaU = (__bf16*)alloc((size_t)DHv * GAT * sizeof(__bf16));
    __bf16* BaV = (__bf16*)alloc((size_t)DHv * GAT * sizeof(__bf16));
    __bf16* BaS = (__bf16*)alloc((size_t)DHv * GAT * sizeof(__bf16));
    float* bias_q0 = (float*)alloc(GAT * sizeof(float));
    float* bias_q1 = (float*)alloc(GAT * sizeof(float));
    float* bias_l  = (float*)alloc(GAT * sizeof(float));
    float* bias_a  = (float*)alloc(GAT * sizeof(float));

    // -------- one-time (per call) prep --------
    {
        const int nS = (int)(11 * ND);   // f32 states
        zero_words_kernel<<<(nS + 255) / 256, 256, 0, stream>>>((unsigned int*)states, nS);
        const int nM = (int)(5 * ND * sizeof(__bf16) / 4);  // bf16 mirrors
        zero_words_kernel<<<(nM + 255) / 256, 256, 0, stream>>>((unsigned int*)mirrors, nM);

        auto pack = [&](const float* src, __bf16* dst, int K, int Kpad, int trans) {
            const int n = (GAT / 16) * (Kpad / 32) * 512;
            pack_weight_kernel<<<(n + 255) / 256, 256, 0, stream>>>(src, dst, K, Kpad, trans);
        };
        pack(q0_wih, Bq0_ih, DHv, DHv, 1);   // (512,128) row-major -> B = w^T
        pack(q0_whh, Bq0_hh, DHv, DHv, 1);
        pack(q1_wih, Bq1_ih, DHv, DHv, 1);
        pack(q1_whh, Bq1_hh, DHv, DHv, 1);
        pack(Wl_w, BlW, DLDIM, KLPAD, 0);    // (K,512) row-major direct
        pack(Ul_w, BlU, DHv, DHv, 0);
        pack(Vl_w, BlV, DHv, DHv, 0);
        pack(Sl_w, BlS, DHv, DHv, 0);
        pack(Wa_w, BaW, DADIM, KAPAD, 0);
        pack(Ua_w, BaU, DHv, DHv, 0);
        pack(Va_w, BaV, DHv, DHv, 0);
        pack(Sa_w, BaS, DHv, DHv, 0);

        combine_bias_kernel<<<2, 256, 0, stream>>>(bias_q0, q0_bih, q0_bhh, nullptr, nullptr);
        combine_bias_kernel<<<2, 256, 0, stream>>>(bias_q1, q1_bih, q1_bhh, nullptr, nullptr);
        combine_bias_kernel<<<2, 256, 0, stream>>>(bias_l, Wl_b, Ul_b, Vl_b, Sl_b);
        combine_bias_kernel<<<2, 256, 0, stream>>>(bias_a, Wa_b, Ua_b, Va_b, Sa_b);
    }

    const int ndThreads = 256;
    const int ndBlocks  = (int)((ND + ndThreads - 1) / ndThreads);

    // -------- sequential scan over timesteps --------
    for (int t = 0; t < TSTEPS; ++t) {
        const float* qm_t = qmask + (size_t)t * NBAT * 2;
        const float* xl_t = x_l + (size_t)t * NBAT * DLDIM;
        const float* xa_t = x_a + (size_t)t * NBAT * DADIM;
        float* out_t = out + (size_t)t * NBAT * GAT;

        perm_kernel<<<1, NBAT, 0, stream>>>(qm_t, qm_idx, perm0, perm1, n0_ptr);
        gather_sel_kernel<<<ndBlocks, ndThreads, 0, stream>>>(
            q, qm_idx, perm0, perm1, q0_sel, q1_sel, q0_sel_b, q1_sel_b);

        {
            const int nL = NBAT * KLPAD;
            convert_x_kernel<<<(nL + 255) / 256, 256, 0, stream>>>(xl_t, xl_b, DLDIM, KLPAD);
            const int nA = NBAT * KAPAD;
            convert_x_kernel<<<(nA + 255) / 256, 256, 0, stream>>>(xa_t, xa_b, DADIM, KAPAD);
        }

        // LSTM cells: gates = sel @ w_ih^T + h @ w_hh^T + (b_ih+b_hh)
        launch_gemm(stream, gates_q0, bias_q0,
                    q0_sel_b, Bq0_ih, DHv, h_q0_b, Bq0_hh, DHv,
                    nullptr, nullptr, 0, nullptr, nullptr, 0, 2);
        launch_gemm(stream, gates_q1, bias_q1,
                    q1_sel_b, Bq1_ih, DHv, h_q1_b, Bq1_hh, DHv,
                    nullptr, nullptr, 0, nullptr, nullptr, 0, 2);
        lstm_gates_kernel<<<ndBlocks, ndThreads, 0, stream>>>(gates_q0, h_q0, c_q0, h_q0_b, n0_ptr, 0);
        lstm_gates_kernel<<<ndBlocks, ndThreads, 0, stream>>>(gates_q1, h_q1, c_q1, h_q1_b, n0_ptr, 1);

        compose_kernel<<<ndBlocks, ndThreads, 0, stream>>>(
            h_q0, h_q1, q0_sel, q1_sel, n0_ptr, qm_t, h_q, h_q_b, q, out_t);

        // LSTHM cells: sums = x@W + h@U + z@V + h_q@S + combined bias
        launch_gemm(stream, sums_l, bias_l,
                    xl_b, BlW, KLPAD, h_l_b, BlU, DHv,
                    z_l_b, BlV, DHv, h_q_b, BlS, DHv, 4);
        launch_gemm(stream, sums_a, bias_a,
                    xa_b, BaW, KAPAD, h_a_b, BaU, DHv,
                    z_l_b, BaV, DHv, h_q_b, BaS, DHv, 4);
        lsthm_gates_kernel<<<ndBlocks, ndThreads, 0, stream>>>(sums_l, c_l, h_l, h_l_b, out_t, 0);
        lsthm_gates_kernel<<<ndBlocks, ndThreads, 0, stream>>>(sums_a, c_a, h_a, h_a_b, out_t, DHv);

        attention_kernel<<<NBAT, DHv, 0, stream>>>(c_l, c_a, Wq, Wk, z_l, z_l_b, out_t);
    }
    (void)in_sizes; (void)n_in; (void)out_size; (void)ws_size;
}
